// StandardCRF_39737037423424
// MI455X (gfx1250) — matrix operand order repaired and checked by reading, not compiled, not run on hardware
//
#include <hip/hip_runtime.h>
#include <hip/hip_bf16.h>

typedef _Float16 v16h __attribute__((ext_vector_type(16)));
typedef float    v8f  __attribute__((ext_vector_type(8)));
typedef float    v4f  __attribute__((ext_vector_type(4)));

#define SEQ 512
#define NT  48
#define AST 52   // LDS alpha row stride in floats: 52*r mod 64 hits distinct 4-bank groups

#define LOG2E 1.4426950408889634f
#define LN2   0.6931471805599453f

// raw v_exp_f32 / v_log_f32 (base-2); compiler keeps hazard handling
#if __has_builtin(__builtin_amdgcn_exp2f)
#define EXP2(x) __builtin_amdgcn_exp2f(x)
#else
#define EXP2(x) exp2f(x)
#endif
#if __has_builtin(__builtin_amdgcn_logf)
#define LOG2(x) __builtin_amdgcn_logf(x)
#else
#define LOG2(x) __log2f(x)
#endif

// ---------------- workspace layout (bytes) ----------------
// [0      .. 4096)  float score[1024]
// [4096   .. 8192)  int   len[1024]
// [8192   .. 8448)  float c2[64]           (column max of transitions, log2 units)
// [8448   ..14592)  _Float16 bfrag[6][32][16]  (WMMA B fragments, lane-major)

// ============================================================
// Kernel 1: prep — column maxes, exp2(trans*log2e - c2) packed into
// the v_wmma 32x16 f16 B-fragment lane layout; zero the output.
// B frag layout (16-bit, wave32): lanes 0-15 hold K=0..7,16..23,
// lanes 16-31 hold K=8..15,24..31 (h -> K=8*half+h, +8 for h>=8); N = lane%16.
// Fragment f = kt*3 + nt covers K tile kt*32.., N tile nt*16..
// ============================================================
__global__ void crf_prep_kernel(const float* __restrict__ trans,
                                float* __restrict__ c_ws,
                                _Float16* __restrict__ bf_ws,
                                float* __restrict__ out) {
  __shared__ float cl[NT];   // column max, log2 units
  const int t = threadIdx.x;
  if (t == 0) out[0] = 0.0f;
  if (t < NT) {
    float mx = trans[t];
    for (int k = 1; k < NT; ++k) mx = fmaxf(mx, trans[k * NT + t]);
    const float c2 = mx * LOG2E;
    cl[t] = c2;
    c_ws[t] = c2;
  }
  __syncthreads();
  if (t < 192) {                       // 6 fragments * 32 lanes
    const int f    = t >> 5;
    const int lane = t & 31;
    const int kt   = f / 3;
    const int nt   = f % 3;
    const int kb   = kt * 32 + ((lane >> 4) << 3);   // K base for this lane
    const int n    = nt * 16 + (lane & 15);
    const float cn = cl[n];
    _Float16* o = bf_ws + t * 16;
#pragma unroll
    for (int h = 0; h < 16; ++h) {
      const int k = kb + h + ((h >> 3) << 3);
      const float v = (k < NT) ? EXP2(fmaf(trans[k * NT + n], LOG2E, -cn)) : 0.0f;
      o[h] = (_Float16)v;
    }
  }
}

// ============================================================
// Kernel 2: gold score + sequence length per batch (mask is a
// contiguous prefix, so len fully determines it).
// ============================================================
__global__ void crf_score_kernel(const float* __restrict__ em,
                                 const int* __restrict__ tags,
                                 const int* __restrict__ mask,
                                 const float* __restrict__ startt,
                                 const float* __restrict__ endt,
                                 const float* __restrict__ trans,
                                 float* __restrict__ score_ws,
                                 int* __restrict__ len_ws,
                                 int nbatch) {
  const int b = blockIdx.x * blockDim.x + threadIdx.x;
  if (b >= nbatch) return;
  const int* tg = tags + (size_t)b * SEQ;
  const int* mk = mask + (size_t)b * SEQ;
  int len = 0;
  for (int i = 0; i < SEQ; ++i) len += mk[i];

  const float* eb = em + (size_t)b * SEQ * NT;
  float sc = startt[tg[0]];
  // emission sum over i in [0, S-2] with mask[i] -> i < min(len, S-1)
  const int eend = (len < SEQ) ? len : (SEQ - 1);
  float es = 0.0f;
  for (int i = 0; i < eend; ++i) es += eb[(size_t)i * NT + tg[i]];
  // transition sum: mask[i+1] -> i+1 < len
  for (int i = 0; i + 1 < len; ++i) sc += trans[tg[i] * NT + tg[i + 1]];
  const int last = tg[len - 1];
  sc += es + eb[(size_t)(len - 1) * NT + last] + endt[last];

  score_ws[b] = sc;
  len_ws[b]   = len;
}

// ============================================================
// Kernel 3: forward algorithm (log2 domain). One wave32 per 16 batches.
// ============================================================
__global__ __launch_bounds__(32) void crf_forward_kernel(
    const float* __restrict__ em,
    const float* __restrict__ startt,
    const float* __restrict__ endt,
    const float* __restrict__ score_ws,
    const int* __restrict__ len_ws,
    const float* __restrict__ c_ws,
    const _Float16* __restrict__ bf_ws,
    float* __restrict__ out,
    float inv_b) {
  __shared__ __align__(16) float alds[16 * AST];  // alphas * log2e, f32 [row][col]
  __shared__ __align__(16) float mlds[16];        // per-row max (log2 units)

  const int lane  = threadIdx.x;
  const int b0    = blockIdx.x * 16;
  const bool lo   = lane < 16;
  const int r     = lane & 15;          // A-row / D-col index
  const int rbase = lo ? 0 : 8;         // row-offset (D)
  const int kb    = lo ? 0 : 8;         // K-offset (A fragment)

  // ---- resident B fragments: exp(trans - colmax), constant for all steps ----
  const v16h* bfp = (const v16h*)bf_ws;
  v16h bf[6];
#pragma unroll
  for (int f = 0; f < 6; ++f) bf[f] = bfp[f * 32 + lane];

  // column-max constants (log2 units) for this lane's D columns
  const float c0 = c_ws[r];
  const float c1 = c_ws[16 + r];
  const float c2 = c_ws[32 + r];

  // lengths for this lane's D rows; tile-wide minimum for the unguarded path
  int lenr[8];
#pragma unroll
  for (int j = 0; j < 8; ++j) lenr[j] = len_ws[b0 + rbase + j];
  int myMin = lenr[0];
#pragma unroll
  for (int j = 1; j < 8; ++j) myMin = min(myMin, lenr[j]);
  myMin = min(myMin, __shfl_xor(myMin, 16, 32));
  const int tileMin = __builtin_amdgcn_readfirstlane(myMin);

  // ---- init alphas: (start + em[:,0,:]) * log2e ----
  for (int e = lane; e < 16 * NT; e += 32) {
    const int row = e / NT;
    const int col = e % NT;
    alds[row * AST + col] =
        (startt[col] + em[(size_t)(b0 + row) * SEQ * NT + col]) * LOG2E;
  }
  // single-wave workgroup: LDS ops are in program order; no barrier needed.

  // per-lane emission base pointer for the D phase (row b0+rbase, col r, i=1);
  // all 24 loads/step use constant immediate offsets off this pointer.
  const float* emb = em + ((size_t)(b0 + rbase) * SEQ + 1) * NT + r;

  for (int i = 1; i < SEQ; ++i) {
    // ---- row phase: lane loads only its 6 chunks of row r ----
    const float* rp = &alds[r * AST];
    const v4f u0 = *(const v4f*)(rp + kb);        // K kb+0..3
    const v4f u1 = *(const v4f*)(rp + kb + 4);    // K kb+4..7
    const v4f u2 = *(const v4f*)(rp + kb + 16);   // K kb+16..19
    const v4f u3 = *(const v4f*)(rp + kb + 20);   // K kb+20..23
    const v4f u4 = *(const v4f*)(rp + kb + 32);   // K kb+32..35
    const v4f u5 = *(const v4f*)(rp + kb + 36);   // K kb+36..39

    // partial row max over this lane's 24 cols; combine across half-waves
#define MX4(Q) pm = fmaxf(pm, fmaxf(fmaxf(Q[0], Q[1]), fmaxf(Q[2], Q[3])))
    float pm = u0[0];
    MX4(u0); MX4(u1); MX4(u2); MX4(u3); MX4(u4); MX4(u5);
#undef MX4
    const float m = fmaxf(pm, __shfl_xor(pm, 16, 32));
    if (lo) mlds[r] = m;

    // ---- build A fragments (16x32 f16 layout), K 48..63 zero-padded ----
    v16h a0, a1;
#pragma unroll
    for (int e = 0; e < 4; ++e) {
      a0[e]      = (_Float16)EXP2(u0[e] - m);
      a0[4 + e]  = (_Float16)EXP2(u1[e] - m);
      a0[8 + e]  = (_Float16)EXP2(u2[e] - m);
      a0[12 + e] = (_Float16)EXP2(u3[e] - m);
      a1[e]      = (_Float16)EXP2(u4[e] - m);
      a1[4 + e]  = (_Float16)EXP2(u5[e] - m);
      a1[8 + e]  = (_Float16)0.0f;
      a1[12 + e] = (_Float16)0.0f;
    }

    // ---- 6 WMMAs: 3 N tiles x 2 K steps, f32 accumulate ----
    v8f d0 = {}, d1 = {}, d2 = {};
    d0 = __builtin_amdgcn_wmma_f32_16x16x32_f16(false, a0, false, bf[0],
                                                (short)0, d0, false, false);
    d1 = __builtin_amdgcn_wmma_f32_16x16x32_f16(false, a0, false, bf[1],
                                                (short)0, d1, false, false);
    d2 = __builtin_amdgcn_wmma_f32_16x16x32_f16(false, a0, false, bf[2],
                                                (short)0, d2, false, false);
    d0 = __builtin_amdgcn_wmma_f32_16x16x32_f16(false, a1, false, bf[3],
                                                (short)0, d0, false, false);
    d1 = __builtin_amdgcn_wmma_f32_16x16x32_f16(false, a1, false, bf[4],
                                                (short)0, d1, false, false);
    d2 = __builtin_amdgcn_wmma_f32_16x16x32_f16(false, a1, false, bf[5],
                                                (short)0, d2, false, false);

    // ---- finish in D layout: lane holds col r, rows rbase..rbase+7 ----
    const v4f ml = *(const v4f*)&mlds[rbase];
    const v4f mh = *(const v4f*)&mlds[rbase + 4];

    if (i < tileMin) {
      // fast path: every row still active -> unconditional stores
#pragma unroll
      for (int j = 0; j < 8; ++j) {
        const int row = rbase + j;
        const float mj = (j < 4) ? ml[j & 3] : mh[j & 3];
        const float* ep = emb + (size_t)j * (SEQ * NT);
        __builtin_prefetch(ep + NT, 0, 3);   // next step's emissions
        alds[row * AST + r]      = fmaf(ep[0],  LOG2E, mj + c0 + LOG2(d0[j]));
        alds[row * AST + 16 + r] = fmaf(ep[16], LOG2E, mj + c1 + LOG2(d1[j]));
        alds[row * AST + 32 + r] = fmaf(ep[32], LOG2E, mj + c2 + LOG2(d2[j]));
      }
    } else {
      // tail: some rows may be frozen (keep old alphas by skipping store)
#pragma unroll
      for (int j = 0; j < 8; ++j) {
        if (i < lenr[j]) {
          const int row = rbase + j;
          const float mj = (j < 4) ? ml[j & 3] : mh[j & 3];
          const float* ep = emb + (size_t)j * (SEQ * NT);
          alds[row * AST + r]      = fmaf(ep[0],  LOG2E, mj + c0 + LOG2(d0[j]));
          alds[row * AST + 16 + r] = fmaf(ep[16], LOG2E, mj + c1 + LOG2(d1[j]));
          alds[row * AST + 32 + r] = fmaf(ep[32], LOG2E, mj + c2 + LOG2(d2[j]));
        }
      }
    }
    emb += NT;
  }

  // ---- log partition + reduction into scalar output ----
  if (lo) {
    float v[NT];
    float mm = -3.402823466e38f;
#pragma unroll
    for (int t = 0; t < NT; ++t) {
      v[t] = alds[r * AST + t] + endt[t] * LOG2E;   // log2 units
      mm = fmaxf(mm, v[t]);
    }
    float s = 0.0f;
#pragma unroll
    for (int t = 0; t < NT; ++t) s += EXP2(v[t] - mm);
    const float lz = (mm + LOG2(s)) * LN2;          // back to nat log
    atomicAdd(out, (lz - score_ws[b0 + r]) * inv_b);
  }
}

// ============================================================
extern "C" void kernel_launch(void* const* d_in, const int* in_sizes, int n_in,
                              void* d_out, int out_size, void* d_ws, size_t ws_size,
                              hipStream_t stream) {
  const float* em     = (const float*)d_in[0];
  const int*   tags   = (const int*)d_in[1];
  const int*   mask   = (const int*)d_in[2];
  const float* startt = (const float*)d_in[3];
  const float* endt   = (const float*)d_in[4];
  const float* trans  = (const float*)d_in[5];
  float* out = (float*)d_out;

  const int nbatch = in_sizes[1] / SEQ;   // tags is [B, S]

  char* ws = (char*)d_ws;
  float*    score_ws = (float*)(ws + 0);
  int*      len_ws   = (int*)(ws + 4096);
  float*    c_ws     = (float*)(ws + 8192);
  _Float16* bf_ws    = (_Float16*)(ws + 8448);

  crf_prep_kernel<<<1, 256, 0, stream>>>(trans, c_ws, bf_ws, out);
  crf_score_kernel<<<(nbatch + 255) / 256, 256, 0, stream>>>(
      em, tags, mask, startt, endt, trans, score_ws, len_ws, nbatch);
  crf_forward_kernel<<<nbatch / 16, 32, 0, stream>>>(
      em, startt, endt, score_ws, len_ws, c_ws, bf_ws, out,
      1.0f / (float)nbatch);
}
